// DynamicSnakeConv_32014686224930
// MI455X (gfx1250) — matrix operand order repaired and checked by reading, not compiled, not run on hardware
//
#include <hip/hip_runtime.h>

// DynamicSnakeConv for MI455X (gfx1250, wave32, WMMA).
//
// Kernel 1: fused conv1->relu->conv2->relu->conv3(time channels only) + the
//           8-step integer-offset scan; writes grid (B,9,L) float to d_ws.
// Kernel 2: gather/lerp of x at grid positions + 64x576 @ 576x64 GEMM per
//           (b, l-tile) via v_wmma_f32_16x16x32_f16 with f32 accumulation.

typedef __attribute__((ext_vector_type(16))) _Float16 v16h;
typedef __attribute__((ext_vector_type(8)))  float    v8f;

#define B_     8
#define C_     64
#define COUT_  64
#define K_     9
#define L_     65536
#define T1     64          // kernel1 l-tile width
#define TL     64          // kernel2 l-tile width
#define CK     576         // contraction length = K_*C_
#define NCHUNK 18          // CK / 32

// ---------------------------------------------------------------------------
// Kernel 1: offsets + scan
// ---------------------------------------------------------------------------
__global__ __launch_bounds__(256) void snake_offsets_kernel(
    const float* __restrict__ x,
    const float* __restrict__ w1, const float* __restrict__ b1,
    const float* __restrict__ w2, const float* __restrict__ b2,
    const float* __restrict__ w3, const float* __restrict__ b3,
    float* __restrict__ grid)
{
    __shared__ float xs [C_][T1 + 6];   // x tile, halo 3 each side
    __shared__ float h1s[32][T1 + 4];   // conv1 out, halo 2
    __shared__ float h2s[32][T1 + 2];   // conv2 out, halo 1
    __shared__ float toffs[8][T1];      // time offsets

    const int tiles = L_ / T1;
    const int b   = blockIdx.x / tiles;
    const int l0  = (blockIdx.x % tiles) * T1;
    const int tid = threadIdx.x;

    // ---- load x tile with zero halo (conv1's zero padding at L boundaries)
    for (int i = tid; i < C_ * (T1 + 6); i += 256) {
        int c = i / (T1 + 6), j = i % (T1 + 6);
        int g = l0 - 3 + j;
        xs[c][j] = (g >= 0 && g < L_) ? x[(b * C_ + c) * L_ + g] : 0.f;
    }
    __syncthreads();

    // ---- conv1 + relu on [l0-2, l0+T1+2); positions outside [0,L) are the
    //      zero padding of conv2, so force 0 there.
    for (int i = tid; i < 32 * (T1 + 4); i += 256) {
        int oc = i / (T1 + 4), p = i % (T1 + 4);
        int g = l0 - 2 + p;
        float acc = 0.f;
        if (g >= 0 && g < L_) {
            acc = b1[oc];
            const float* wr = w1 + oc * C_ * 3;
            #pragma unroll 4
            for (int ic = 0; ic < C_; ++ic) {
                acc = fmaf(wr[ic * 3 + 0], xs[ic][p + 0], acc);
                acc = fmaf(wr[ic * 3 + 1], xs[ic][p + 1], acc);
                acc = fmaf(wr[ic * 3 + 2], xs[ic][p + 2], acc);
            }
            acc = fmaxf(acc, 0.f);
        }
        h1s[oc][p] = acc;
    }
    __syncthreads();

    // ---- conv2 + relu on [l0-1, l0+T1+1)
    for (int i = tid; i < 32 * (T1 + 2); i += 256) {
        int oc = i / (T1 + 2), p = i % (T1 + 2);
        int g = l0 - 1 + p;
        float acc = 0.f;
        if (g >= 0 && g < L_) {
            acc = b2[oc];
            const float* wr = w2 + oc * 32 * 3;
            #pragma unroll 4
            for (int ic = 0; ic < 32; ++ic) {
                acc = fmaf(wr[ic * 3 + 0], h1s[ic][p + 0], acc);
                acc = fmaf(wr[ic * 3 + 1], h1s[ic][p + 1], acc);
                acc = fmaf(wr[ic * 3 + 2], h1s[ic][p + 2], acc);
            }
            acc = fmaxf(acc, 0.f);
        }
        h2s[oc][p] = acc;
    }
    __syncthreads();

    // ---- conv3: only odd output channels (time offsets) are ever used
    for (int i = tid; i < 8 * T1; i += 256) {
        int kk = i / T1, p = i % T1;
        int oc = 2 * kk + 1;
        float acc = b3[oc];
        const float* wr = w3 + oc * 32 * 3;
        #pragma unroll 4
        for (int ic = 0; ic < 32; ++ic) {
            acc = fmaf(wr[ic * 3 + 0], h2s[ic][p + 0], acc);
            acc = fmaf(wr[ic * 3 + 1], h2s[ic][p + 1], acc);
            acc = fmaf(wr[ic * 3 + 2], h2s[ic][p + 2], acc);
        }
        toffs[kk][p] = acc;
    }
    __syncthreads();

    // ---- sequential scan over k (local per l): g += sign(t)*ceil(|t|), clip
    if (tid < T1) {
        int l = l0 + tid;
        float g = (float)l;
        grid[(b * K_ + 0) * L_ + l] = g;
        #pragma unroll
        for (int k = 1; k < K_; ++k) {
            float t  = toffs[k - 1][tid];
            float ts = (t > 0.f) ? ceilf(t) : ((t < 0.f) ? -ceilf(-t) : 0.f);
            g = fminf(fmaxf(g + ts, 0.f), (float)(L_ - 1));
            grid[(b * K_ + k) * L_ + l] = g;
        }
    }
}

// ---------------------------------------------------------------------------
// Kernel 2: gather + WMMA GEMM
//   out[b,o,l] = bias[o] + sum_{ck} weight[o, ck] * sampled[b, ck, l]
//   ck = k*64 + c;  sampled = w0*x[b,c,p0[k,l]] + w1*x[b,c,p1[k,l]]
// Workgroup: 64 o x 64 l tile, 8 waves; wave w -> osub=w>>1, lsubs {2*(w&1), 2*(w&1)+1}
// ---------------------------------------------------------------------------
__global__ __launch_bounds__(256) void snake_gemm_kernel(
    const float* __restrict__ x,
    const float* __restrict__ weight,
    const float* __restrict__ bias,
    const float* __restrict__ grid,
    float* __restrict__ out)
{
    // Fragment-ordered staging: consumer lane L reads halfs [L*16, L*16+16).
    __shared__ __align__(32) _Float16 ldsA[4 * 512];  // 4 osubs x (32 lanes x 16 halfs)
    __shared__ __align__(32) _Float16 ldsB[4 * 512];  // 4 lsubs x (32 lanes x 16 halfs)
    __shared__ int   s_p0[CK];
    __shared__ float s_w0[CK];
    __shared__ float s_w1[CK];

    const int tiles = L_ / TL;
    const int b    = blockIdx.x / tiles;
    const int l0   = (blockIdx.x % tiles) * TL;
    const int tid  = threadIdx.x;
    const int wave = tid >> 5;
    const int lane = tid & 31;

    // ---- per-(k, l_local) sampling positions & lerp weights
    for (int i = tid; i < CK; i += 256) {
        int k = i >> 6, ll = i & 63;
        float g = grid[(b * K_ + k) * L_ + l0 + ll];
        int p0 = (int)floorf(g);
        p0 = max(0, min(p0, L_ - 1));
        int p1 = min(p0 + 1, L_ - 1);
        s_p0[i] = p0;
        s_w0[i] = (float)p1 - g;   // wt0 = pos1 - grid
        s_w1[i] = g - (float)p0;   // wt1 = grid - pos0
    }

    const int osub = wave >> 1;
    const int lp   = wave & 1;
    const float* xb = x + (b * C_) * L_;

    v8f acc0 = {0.f, 0.f, 0.f, 0.f, 0.f, 0.f, 0.f, 0.f};
    v8f acc1 = {0.f, 0.f, 0.f, 0.f, 0.f, 0.f, 0.f, 0.f};

    for (int t = 0; t < NCHUNK; ++t) {
        __syncthreads();   // protect previous chunk's consumers (and pos precompute)

        // ---- stage A chunk (64 o x 32 ck) in WMMA A fragment order:
        // lane<16: K = j (j<8) else j+8 ; lane>=16: K = j+8 (j<8) else j+16
        for (int i = tid; i < 2048; i += 256) {
            int os = i >> 9, r = i & 511, fl = r >> 4, j = r & 15;
            int m  = fl & 15;
            int Kk = (fl < 16) ? ((j < 8) ? j : j + 8)
                               : ((j < 8) ? j + 8 : j + 16);
            int ck = 32 * t + Kk;
            int c = ck & 63, kq = ck >> 6;
            ldsA[i] = (_Float16)weight[((os * 16 + m) * C_ + c) * K_ + kq];
        }
        // ---- stage B chunk (32 ck x 64 l) in WMMA B fragment order:
        // lane<16: K = j (j<8) else j+8 ; lane>=16: K = j+8 (j<8) else j+16 ; N = lane&15
        for (int i = tid; i < 2048; i += 256) {
            int ls = i >> 9, r = i & 511, fl = r >> 4, j = r & 15;
            int n  = fl & 15;
            int Kk = (fl < 16) ? ((j < 8) ? j : j + 8)
                               : ((j < 8) ? j + 8 : j + 16);
            int ck = 32 * t + Kk;
            int c = ck & 63, kq = ck >> 6;
            int pi = (kq << 6) + (ls * 16 + n);
            int p0 = s_p0[pi];
            int p1 = min(p0 + 1, L_ - 1);
            const float* xc = xb + c * L_;
            float v = s_w0[pi] * xc[p0] + s_w1[pi] * xc[p1];
            ldsB[i] = (_Float16)v;
        }
        __syncthreads();

        v16h a  = *(const v16h*)&ldsA[osub * 512 + lane * 16];
        v16h b0 = *(const v16h*)&ldsB[(lp * 2 + 0) * 512 + lane * 16];
        v16h b1 = *(const v16h*)&ldsB[(lp * 2 + 1) * 512 + lane * 16];
        acc0 = __builtin_amdgcn_wmma_f32_16x16x32_f16(false, a, false, b0,
                                                      (short)0, acc0, false, false);
        acc1 = __builtin_amdgcn_wmma_f32_16x16x32_f16(false, a, false, b1,
                                                      (short)0, acc1, false, false);
    }

    // ---- store D (f32 16x16 layout: VGPR r -> M=r (lanes 0-15) / M=r+8 (16-31))
    const int n = lane & 15;
    #pragma unroll
    for (int r = 0; r < 8; ++r) {
        int m = (lane < 16) ? r : r + 8;
        int o = osub * 16 + m;
        float bv = bias[o];
        int la = l0 + (lp * 2 + 0) * 16 + n;
        int lb = l0 + (lp * 2 + 1) * 16 + n;
        out[(b * COUT_ + o) * L_ + la] = acc0[r] + bv;
        out[(b * COUT_ + o) * L_ + lb] = acc1[r] + bv;
    }
}

// ---------------------------------------------------------------------------
extern "C" void kernel_launch(void* const* d_in, const int* in_sizes, int n_in,
                              void* d_out, int out_size, void* d_ws, size_t ws_size,
                              hipStream_t stream) {
    const float* x      = (const float*)d_in[0];
    const float* w1     = (const float*)d_in[1];
    const float* b1     = (const float*)d_in[2];
    const float* w2     = (const float*)d_in[3];
    const float* b2     = (const float*)d_in[4];
    const float* w3     = (const float*)d_in[5];
    const float* b3     = (const float*)d_in[6];
    const float* weight = (const float*)d_in[7];
    const float* bias   = (const float*)d_in[8];
    float* out  = (float*)d_out;
    float* grid = (float*)d_ws;   // B*9*L floats = 18.9 MB of scratch

    (void)in_sizes; (void)n_in; (void)out_size; (void)ws_size;

    snake_offsets_kernel<<<B_ * (L_ / T1), 256, 0, stream>>>(
        x, w1, b1, w2, b2, w3, b3, grid);
    snake_gemm_kernel<<<B_ * (L_ / TL), 256, 0, stream>>>(
        x, weight, bias, grid, out);
}
